// MoE_82832739271216
// MI455X (gfx1250) — matrix operand order repaired and checked by reading, not compiled, not run on hardware
//
#include <hip/hip_runtime.h>
#include <hip/hip_bf16.h>

#define DIM   2048
#define HID   1024
#define NE    8
#define TOPK  2
#define NTOK  2048
#define NSLOT (NTOK * TOPK)

#define BM 64
#define BN 128
#define BK 32
#define LDA 40   // LDS row stride in ushorts (80B: 16B-aligned, bank-spread)

typedef __attribute__((ext_vector_type(16))) __bf16          bf16x16;
typedef __attribute__((ext_vector_type(8)))  float           floatx8;
typedef __attribute__((ext_vector_type(8)))  unsigned short  ushortx8;
typedef __attribute__((ext_vector_type(4)))  unsigned short  ushortx4;

// fp32 -> bf16 via compiler-native conversion (backend emits v_cvt_pk_bf16_f32)
__device__ __forceinline__ unsigned short f2bf(float f) {
  return __builtin_bit_cast(unsigned short, (__bf16)f);
}
__device__ __forceinline__ ushortx4 f4bf4(float a, float b, float c, float d) {
  ushortx4 r;
  r[0] = f2bf(a); r[1] = f2bf(b); r[2] = f2bf(c); r[3] = f2bf(d);
  return r;
}

// Fast sigmoid: v_exp_f32 + v_rcp_f32 (adequate next to bf16 GEMM precision)
__device__ __forceinline__ float fast_sigmoid(float v) {
  return __builtin_amdgcn_rcpf(1.f + __expf(-v));
}

// Build a 16-element bf16 fragment from two 8-element LDS chunks
__device__ __forceinline__ bf16x16 frag16(const unsigned short* p0, const unsigned short* p1) {
  ushortx8 lo = *(const ushortx8*)p0;
  ushortx8 hi = *(const ushortx8*)p1;
  return __builtin_bit_cast(bf16x16,
      __builtin_shufflevector(lo, hi, 0,1,2,3,4,5,6,7,8,9,10,11,12,13,14,15));
}

__device__ __forceinline__ floatx8 wmma_bf16(bf16x16 a, bf16x16 b, floatx8 c) {
  return __builtin_amdgcn_wmma_f32_16x16x32_bf16(false, a, false, b, (short)0, c, false, false);
}

// ---------------- 1. Router: sigmoid scores + biased top-2 ----------------
__global__ __launch_bounds__(256) void router_kernel(
    const float* __restrict__ x, const float* __restrict__ gw,
    const float* __restrict__ bias,
    int* __restrict__ flat_exp, float* __restrict__ top_sc) {
  int lane = threadIdx.x & 31;
  int wave = threadIdx.x >> 5;
  int t = blockIdx.x * 8 + wave;
  if (t >= NTOK) return;
  float acc[NE];
  #pragma unroll
  for (int e = 0; e < NE; ++e) acc[e] = 0.f;
  const float* xr = x + (size_t)t * DIM;
  for (int k = lane; k < DIM; k += 32) {
    float xv = xr[k];
    #pragma unroll
    for (int e = 0; e < NE; ++e) acc[e] += xv * gw[e * DIM + k];
  }
  #pragma unroll
  for (int e = 0; e < NE; ++e)
    #pragma unroll
    for (int off = 16; off > 0; off >>= 1) acc[e] += __shfl_xor(acc[e], off, 32);
  if (lane == 0) {
    float s[NE], sb[NE];
    #pragma unroll
    for (int e = 0; e < NE; ++e) { s[e] = fast_sigmoid(acc[e]); sb[e] = s[e] + bias[e]; }
    int b1 = 0;
    #pragma unroll
    for (int e = 1; e < NE; ++e) if (sb[e] > sb[b1]) b1 = e;
    int b2 = (b1 == 0) ? 1 : 0;
    #pragma unroll
    for (int e = 0; e < NE; ++e) if (e != b1 && sb[e] > sb[b2]) b2 = e;
    flat_exp[t * 2 + 0] = b1;  flat_exp[t * 2 + 1] = b2;
    top_sc[t * 2 + 0] = s[b1]; top_sc[t * 2 + 1] = s[b2];
  }
}

// ---------------- 2. Stable counting sort of slots by expert ----------------
#define SORT_T 256
#define CHUNK  (NSLOT / SORT_T)
__global__ __launch_bounds__(SORT_T) void sort_kernel(
    const int* __restrict__ flat_exp, const float* __restrict__ top_sc,
    int* __restrict__ tok_sorted, float* __restrict__ sc_sorted,
    int* __restrict__ slot_pos, int* __restrict__ seg) {
  __shared__ int hist[NE][SORT_T];
  int tid = threadIdx.x;
  int cnt[NE];
  #pragma unroll
  for (int e = 0; e < NE; ++e) cnt[e] = 0;
  int base = tid * CHUNK;
  for (int i = 0; i < CHUNK; ++i) cnt[flat_exp[base + i]]++;
  #pragma unroll
  for (int e = 0; e < NE; ++e) hist[e][tid] = cnt[e];
  __syncthreads();
  if (tid == 0) {
    int run = 0;
    for (int e = 0; e < NE; ++e) {
      seg[e] = run;
      for (int t2 = 0; t2 < SORT_T; ++t2) { int c = hist[e][t2]; hist[e][t2] = run; run += c; }
    }
    seg[NE] = run;  // == NSLOT
  }
  __syncthreads();
  int pos[NE];
  #pragma unroll
  for (int e = 0; e < NE; ++e) pos[e] = hist[e][tid];
  for (int i = 0; i < CHUNK; ++i) {
    int slot = base + i;
    int e = flat_exp[slot];
    int p = pos[e]++;
    tok_sorted[p] = slot / TOPK;
    sc_sorted[p]  = top_sc[slot];
    slot_pos[slot] = p;
  }
}

// ---------- 3. Grouped SwiGLU up-projection: h = silu(A@W1^T)*(A@W3^T) ----------
// z in [0..7]: routed expert z (A = score-scaled gathered tokens); z==8: shared expert.
__global__ __launch_bounds__(256) void swiglu_kernel(
    const float* __restrict__ x, const int* __restrict__ tok_sorted,
    const float* __restrict__ sc_sorted, const int* __restrict__ seg,
    const float* __restrict__ w1, const float* __restrict__ w3,
    const float* __restrict__ sw1, const float* __restrict__ sw3,
    unsigned short* __restrict__ h_routed, unsigned short* __restrict__ h_shared) {
  int z = blockIdx.z;
  bool se = (z == NE);
  int segBase = se ? 0 : seg[z];
  int rowcnt  = se ? NTOK : (seg[z + 1] - segBase);
  int row0 = blockIdx.y * BM;
  if (row0 >= rowcnt) return;
  int col0 = blockIdx.x * BN;
  const float* W1 = se ? sw1 : (w1 + (size_t)z * HID * DIM);
  const float* W3 = se ? sw3 : (w3 + (size_t)z * HID * DIM);

  __shared__ unsigned short As[BM * LDA];
  __shared__ unsigned short B1s[BN * LDA];
  __shared__ unsigned short B3s[BN * LDA];

  int tid = threadIdx.x;
  int lane = tid & 31, wave = tid >> 5;
  int wm = wave & 1, wn = wave >> 1;    // 2 x 4 wave grid
  int lh = lane >> 4;                   // lane half (0/1)
  int l16 = lane & 15;

  floatx8 acc1[2][2], acc3[2][2];
  floatx8 zero8 = {0.f,0.f,0.f,0.f,0.f,0.f,0.f,0.f};
  #pragma unroll
  for (int mi = 0; mi < 2; ++mi)
    #pragma unroll
    for (int ni = 0; ni < 2; ++ni) { acc1[mi][ni] = zero8; acc3[mi][ni] = zero8; }

  // Per-thread A-loader tasks (2): clamped row index, scale->0 for pad rows.
  // Addresses are always valid -> pure global_load path (no flat).
  size_t aoff[2]; float ascale[2]; int ar[2], akq[2];
  #pragma unroll
  for (int tt = 0; tt < 2; ++tt) {
    int t = tid + tt * 256;
    ar[tt] = t >> 3; akq[tt] = t & 7;
    int ri = row0 + ar[tt];
    bool valid = (ri < rowcnt);
    int rc = valid ? ri : (rowcnt - 1);
    if (se) {
      aoff[tt] = (size_t)rc * DIM;
      ascale[tt] = valid ? 1.f : 0.f;
    } else {
      int slot = segBase + rc;
      aoff[tt] = (size_t)tok_sorted[slot] * DIM;
      ascale[tt] = valid ? sc_sorted[slot] : 0.f;
    }
  }

  for (int k0 = 0; k0 < DIM; k0 += BK) {
    // Stage A tile (gather + scale + fp32->bf16), one ds_store_b64 per quad
    #pragma unroll
    for (int tt = 0; tt < 2; ++tt) {
      float4 v = *(const float4*)(x + aoff[tt] + k0 + akq[tt] * 4);
      float s = ascale[tt];
      *(ushortx4*)&As[ar[tt] * LDA + akq[tt] * 4] = f4bf4(v.x * s, v.y * s, v.z * s, v.w * s);
    }
    // Stage B tiles for w1 and w3 (fp32->bf16 on the fly)
    #pragma unroll
    for (int i = 0; i < 4; ++i) {
      int t = tid + i * 256;
      int n = t >> 3, kq = t & 7;
      size_t src = (size_t)(col0 + n) * DIM + k0 + kq * 4;
      float4 v1 = *(const float4*)(W1 + src);
      float4 v3 = *(const float4*)(W3 + src);
      *(ushortx4*)&B1s[n * LDA + kq * 4] = f4bf4(v1.x, v1.y, v1.z, v1.w);
      *(ushortx4*)&B3s[n * LDA + kq * 4] = f4bf4(v3.x, v3.y, v3.z, v3.w);
    }
    __syncthreads();

    // WMMA: A frags per ISA 16-bit A layout; B frags per 16-bit B layout
    bf16x16 af[2];
    #pragma unroll
    for (int mi = 0; mi < 2; ++mi) {
      int r = wm * 32 + mi * 16 + l16;
      int kb = lh * 8;
      af[mi] = frag16(&As[r * LDA + kb], &As[r * LDA + kb + 16]);
    }
    #pragma unroll
    for (int ni = 0; ni < 2; ++ni) {
      int n = wn * 32 + ni * 16 + l16;
      int kb2 = lh * 8;
      bf16x16 b1 = frag16(&B1s[n * LDA + kb2], &B1s[n * LDA + kb2 + 16]);
      bf16x16 b3 = frag16(&B3s[n * LDA + kb2], &B3s[n * LDA + kb2 + 16]);
      #pragma unroll
      for (int mi = 0; mi < 2; ++mi) {
        acc1[mi][ni] = wmma_bf16(af[mi], b1, acc1[mi][ni]);
        acc3[mi][ni] = wmma_bf16(af[mi], b3, acc3[mi][ni]);
      }
    }
    __syncthreads();
  }

  // Epilogue: h = silu(s1) * s3 = s1*sigmoid(s1)*s3, store bf16 per C-layout
  #pragma unroll
  for (int mi = 0; mi < 2; ++mi)
    #pragma unroll
    for (int ni = 0; ni < 2; ++ni) {
      int n = col0 + wn * 32 + ni * 16 + l16;
      #pragma unroll
      for (int v = 0; v < 8; ++v) {
        int r = wm * 32 + mi * 16 + v + 8 * lh;
        int ri = row0 + r;
        if (ri < rowcnt) {
          float s1 = acc1[mi][ni][v];
          float s3 = acc3[mi][ni][v];
          float hv = s1 * fast_sigmoid(s1) * s3;
          unsigned short hb = f2bf(hv);
          if (se) h_shared[(size_t)ri * HID + n] = hb;
          else    h_routed[(size_t)(segBase + ri) * HID + n] = hb;
        }
      }
    }
}

// ---------- 4. Down projection: out = h @ W2^T ----------
// SHARED: h_shared [NTOK x HID] -> plain store of full d_out.
// routed: per-expert segment of h_routed -> per-slot fp32 buffer (deterministic).
template <bool SHARED>
__global__ __launch_bounds__(256) void down_kernel(
    const unsigned short* __restrict__ h, const int* __restrict__ seg,
    const float* __restrict__ w2, float* __restrict__ out) {
  int z = SHARED ? 0 : blockIdx.z;
  int segBase = SHARED ? 0 : seg[z];
  int rowcnt  = SHARED ? NTOK : (seg[z + 1] - segBase);
  int row0 = blockIdx.y * BM;
  if (row0 >= rowcnt) return;
  int col0 = blockIdx.x * BN;
  const float* W2 = SHARED ? w2 : (w2 + (size_t)z * DIM * HID);

  __shared__ unsigned short As[BM * LDA];
  __shared__ unsigned short Bs[BN * LDA];

  int tid = threadIdx.x;
  int lane = tid & 31, wave = tid >> 5;
  int wm = wave & 1, wn = wave >> 1;
  int lh = lane >> 4;
  int l16 = lane & 15;

  floatx8 acc[2][2];
  floatx8 zero8 = {0.f,0.f,0.f,0.f,0.f,0.f,0.f,0.f};
  #pragma unroll
  for (int mi = 0; mi < 2; ++mi)
    #pragma unroll
    for (int ni = 0; ni < 2; ++ni) acc[mi][ni] = zero8;

  // Clamped-address A loader; pad rows get value-select to zero (no flat path)
  size_t hoff[2]; bool hvalid[2]; int ar[2], akq[2];
  #pragma unroll
  for (int tt = 0; tt < 2; ++tt) {
    int t = tid + tt * 256;
    ar[tt] = t >> 3; akq[tt] = t & 7;
    int ri = row0 + ar[tt];
    hvalid[tt] = (ri < rowcnt);
    int rc = hvalid[tt] ? ri : (rowcnt - 1);
    hoff[tt] = (size_t)(segBase + rc) * HID;
  }
  ushortx4 zero4 = {0, 0, 0, 0};

  for (int k0 = 0; k0 < HID; k0 += BK) {
    #pragma unroll
    for (int tt = 0; tt < 2; ++tt) {
      ushortx4 v = *(const ushortx4*)(h + hoff[tt] + k0 + akq[tt] * 4);
      *(ushortx4*)&As[ar[tt] * LDA + akq[tt] * 4] = hvalid[tt] ? v : zero4;
    }
    #pragma unroll
    for (int i = 0; i < 4; ++i) {
      int t = tid + i * 256;
      int n = t >> 3, kq = t & 7;
      float4 v = *(const float4*)(W2 + (size_t)(col0 + n) * HID + k0 + kq * 4);
      *(ushortx4*)&Bs[n * LDA + kq * 4] = f4bf4(v.x, v.y, v.z, v.w);
    }
    __syncthreads();

    bf16x16 af[2];
    #pragma unroll
    for (int mi = 0; mi < 2; ++mi) {
      int r = wm * 32 + mi * 16 + l16;
      int kb = lh * 8;
      af[mi] = frag16(&As[r * LDA + kb], &As[r * LDA + kb + 16]);
    }
    #pragma unroll
    for (int ni = 0; ni < 2; ++ni) {
      int n = wn * 32 + ni * 16 + l16;
      int kb2 = lh * 8;
      bf16x16 bf = frag16(&Bs[n * LDA + kb2], &Bs[n * LDA + kb2 + 16]);
      #pragma unroll
      for (int mi = 0; mi < 2; ++mi) acc[mi][ni] = wmma_bf16(af[mi], bf, acc[mi][ni]);
    }
    __syncthreads();
  }

  #pragma unroll
  for (int mi = 0; mi < 2; ++mi)
    #pragma unroll
    for (int ni = 0; ni < 2; ++ni) {
      int n = col0 + wn * 32 + ni * 16 + l16;
      #pragma unroll
      for (int v = 0; v < 8; ++v) {
        int r = wm * 32 + mi * 16 + v + 8 * lh;
        int ri = row0 + r;
        if (ri < rowcnt) {
          float val = acc[mi][ni][v];
          out[(size_t)(SHARED ? ri : (segBase + ri)) * DIM + n] = val;
        }
      }
    }
}

// ---------- 5. Deterministic combine: out[token] += its 2 slot contributions ----------
__global__ __launch_bounds__(256) void combine_kernel(
    const float* __restrict__ outslot, const int* __restrict__ slot_pos,
    float* __restrict__ out) {
  int t = blockIdx.y;
  int d = blockIdx.x * 256 + threadIdx.x;
  int p0 = slot_pos[t * TOPK + 0];
  int p1 = slot_pos[t * TOPK + 1];
  size_t o = (size_t)t * DIM + d;
  out[o] += outslot[(size_t)p0 * DIM + d] + outslot[(size_t)p1 * DIM + d];
}

extern "C" void kernel_launch(void* const* d_in, const int* in_sizes, int n_in,
                              void* d_out, int out_size, void* d_ws, size_t ws_size,
                              hipStream_t stream) {
  (void)in_sizes; (void)n_in; (void)out_size; (void)ws_size;
  const float* x    = (const float*)d_in[0];
  const float* bias = (const float*)d_in[1];
  const float* gw   = (const float*)d_in[2];
  const float* w1   = (const float*)d_in[3];
  const float* w2   = (const float*)d_in[4];
  const float* w3   = (const float*)d_in[5];
  const float* sw1  = (const float*)d_in[6];
  const float* sw2  = (const float*)d_in[7];
  const float* sw3  = (const float*)d_in[8];
  float* out = (float*)d_out;

  char* ws = (char*)d_ws;
  int*   flat_exp   = (int*)  (ws + 0);
  float* top_sc     = (float*)(ws + 16384);
  int*   tok_sorted = (int*)  (ws + 32768);
  float* sc_sorted  = (float*)(ws + 49152);
  int*   slot_pos   = (int*)  (ws + 65536);
  int*   seg        = (int*)  (ws + 81920);
  unsigned short* h_routed = (unsigned short*)(ws + 98304);
  unsigned short* h_shared = h_routed + (size_t)NSLOT * HID;
  float* outslot = (float*)(ws + 98304 + (size_t)(NSLOT + NTOK) * HID * 2);

  router_kernel<<<NTOK / 8, 256, 0, stream>>>(x, gw, bias, flat_exp, top_sc);
  sort_kernel<<<1, SORT_T, 0, stream>>>(flat_exp, top_sc, tok_sorted, sc_sorted, slot_pos, seg);
  swiglu_kernel<<<dim3(HID / BN, NSLOT / BM, NE + 1), 256, 0, stream>>>(
      x, tok_sorted, sc_sorted, seg, w1, w3, sw1, sw3, h_routed, h_shared);
  down_kernel<true><<<dim3(DIM / BN, NTOK / BM), 256, 0, stream>>>(h_shared, seg, sw2, out);
  down_kernel<false><<<dim3(DIM / BN, NSLOT / BM, NE), 256, 0, stream>>>(h_routed, seg, w2, outslot);
  combine_kernel<<<dim3(DIM / 256, NTOK), 256, 0, stream>>>(outslot, slot_pos, out);
}
